// MultiheadAttention_19971597927002
// MI455X (gfx1250) — compile-verified, matrix operand order checked by reading, not run on hardware
//
#include <hip/hip_runtime.h>

#define EMB  768
#define EMB3 2304
#define NH   12
#define HD   64
#define BB   8
#define TT   1024
#define NTOK (BB * TT)   // 8192

typedef __attribute__((ext_vector_type(16))) __bf16       v16bf;
typedef __attribute__((ext_vector_type(8)))  float        v8f;
typedef __attribute__((ext_vector_type(4)))  unsigned int u32x4;
typedef __attribute__((ext_vector_type(8)))  unsigned int u32x8;
typedef __attribute__((ext_vector_type(4)))  float        f32x4;

union FragU {
    struct { u32x4 lo, hi; } q;
    v16bf f;
};

__device__ __forceinline__ v16bf make_frag(u32x4 lo, u32x4 hi) {
    FragU u; u.q.lo = lo; u.q.hi = hi; return u.f;
}

// Load one 16x32 bf16 WMMA fragment (A layout; B uses the same pattern with
// "row" = output column) from a [16 rows][stride] K-contiguous bf16 tile.
__device__ __forceinline__ v16bf load_frag(const __bf16* tile, int stride, int k0) {
    const int lane = threadIdx.x & 31;
    const int r    = lane & 15;
    const int hi   = lane >> 4;
    const __bf16* p = tile + r * stride + k0 + hi * 8;
    u32x4 a = *(const u32x4*)(p);
    u32x4 b = *(const u32x4*)(p + 16);
    return make_frag(a, b);
}

__device__ __forceinline__ v8f wmma_bf16(v16bf a, v16bf b, v8f c) {
    return __builtin_amdgcn_wmma_f32_16x16x32_bf16(
        false, a, false, b, (short)0, c, false, false);
}

// 1-D Tensor Data Mover copy: `elems` bf16 elements, global -> LDS.
// D# group0: count=1 | lds_addr | global_addr[56:0] | type=2.
// D# group1: data_size=2B, tensor_dim0 = tile_dim0 = elems (1-D tile).
__device__ __forceinline__ void tdm_load_1d(unsigned lds_off,
                                            unsigned long long gaddr,
                                            unsigned elems) {
    u32x4 g0;
    g0[0] = 1u;                                  // count=1, user descriptor
    g0[1] = lds_off;                             // lds_addr (bytes)
    g0[2] = (unsigned)gaddr;                     // global_addr[31:0]
    g0[3] = (unsigned)((gaddr >> 32) & 0x01FFFFFFull) | 0x80000000u; // type=2
    u32x8 g1;
    g1[0] = 0x00010000u;                         // wg_mask=0, data_size=1 (2B)
    g1[1] = (elems & 0xFFFFu) << 16;             // tensor_dim0[15:0]
    g1[2] = (elems >> 16) & 0xFFFFu;             // tensor_dim0[31:16], tensor_dim1=0
    g1[3] = (elems & 0xFFFFu) << 16;             // tile_dim0 = elems
    g1[4] = 0u;                                  // tile_dim1=0, tile_dim2=0
    g1[5] = elems;                               // tensor_dim0_stride[31:0]
    g1[6] = 0u;
    g1[7] = 0u;
    asm volatile("tensor_load_to_lds %0, %1" :: "s"(g0), "s"(g1) : "memory");
}

// ---------------------------------------------------------------- fp32 -> bf16
__global__ __launch_bounds__(256) void k_cvt(const float* __restrict__ s,
                                             __bf16* __restrict__ d, int n4) {
    int i = blockIdx.x * 256 + threadIdx.x;
    if (i >= n4) return;
    f32x4 f = ((const f32x4*)s)[i];
    union { __bf16 o[4]; unsigned long long u; } pk;
    pk.o[0] = (__bf16)f[0];
    pk.o[1] = (__bf16)f[1];
    pk.o[2] = (__bf16)f[2];
    pk.o[3] = (__bf16)f[3];
    ((unsigned long long*)d)[i] = pk.u;
}

// ------------------------------------------------------------------- QKV GEMM
// [8192,768](bf16) @ [768,2304](bf16) + bias, scattered into head-major
// Q/K/V [B,NH,T,64] bf16.  Block tile 128x128, BK=32, 8 waves (4x2).
__global__ __launch_bounds__(256) void k_qkv(
    const __bf16* __restrict__ X, const __bf16* __restrict__ W,
    const float* __restrict__ bias,
    __bf16* __restrict__ Qh, __bf16* __restrict__ Kh, __bf16* __restrict__ Vh)
{
    __shared__ __bf16 As[128 * 32];   // [m][k]
    __shared__ __bf16 Bs[128 * 32];   // [n][k] (transposed from W)
    const int tid  = threadIdx.x;
    const int lane = tid & 31, wave = tid >> 5;
    const int wm = wave & 3, wn = wave >> 2;
    const int bm0 = blockIdx.x * 128, bn0 = blockIdx.y * 128;
    const int r = lane & 15, hi = lane >> 4;

    v8f zero = {};
    v8f acc[2][4];
    #pragma unroll
    for (int i = 0; i < 2; ++i)
        #pragma unroll
        for (int j = 0; j < 4; ++j) acc[i][j] = zero;

    for (int k0 = 0; k0 < EMB; k0 += 32) {
        {   // A tile: 128 rows x 32 k, straight copy (K-contiguous already)
            int ar = tid >> 1, ah = tid & 1;
            const u32x4* g = (const u32x4*)(X + (size_t)(bm0 + ar) * EMB + k0 + ah * 16);
            u32x4 a0 = g[0], a1 = g[1];
            u32x4* sp = (u32x4*)(As + ar * 32 + ah * 16);
            sp[0] = a0; sp[1] = a1;
        }
        #pragma unroll
        for (int t2 = 0; t2 < 2; ++t2) {  // B tile with transpose: Bs[n][k] = W[k][n]
            int tsk = tid + t2 * 256;
            int nc = tsk & 15, kr = tsk >> 4;
            u32x4 dv = *(const u32x4*)(W + (size_t)(k0 + kr) * EMB3 + bn0 + nc * 8);
            const __bf16* e = (const __bf16*)&dv;
            #pragma unroll
            for (int j = 0; j < 8; ++j)
                Bs[(nc * 8 + j) * 32 + kr] = e[j];
        }
        __syncthreads();
        v16bf af[2], bfr[4];
        #pragma unroll
        for (int i = 0; i < 2; ++i)
            af[i] = load_frag(As + (wm * 32 + i * 16) * 32, 32, 0);
        #pragma unroll
        for (int j = 0; j < 4; ++j)
            bfr[j] = load_frag(Bs + (wn * 64 + j * 16) * 32, 32, 0);
        #pragma unroll
        for (int i = 0; i < 2; ++i)
            #pragma unroll
            for (int j = 0; j < 4; ++j)
                acc[i][j] = wmma_bf16(af[i], bfr[j], acc[i][j]);
        __syncthreads();
    }

    // Epilogue: + bias, scatter to head-major bf16 Q/K/V.
    // Wave's 64-wide column tile sits in ONE section and ONE head (64-aligned,
    // 768%64==0); the 128-row M tile sits in ONE batch (1024%128==0).
    const int colbase = bn0 + wn * 64;          // multiple of 64
    const int sec = colbase / EMB;              // uniform: 0=Q, 1=K, 2=V
    const int cc0 = colbase - sec * EMB;        // multiple of 64
    const int hh  = cc0 >> 6;                   // uniform head
    const int bidx = bm0 >> 10;                 // uniform batch
    const int t0   = bm0 & 1023;                // row base within T
    __bf16* dst = (sec == 0 ? Qh : (sec == 1 ? Kh : Vh))
                + ((size_t)(bidx * NH + hh)) * TT * HD;
    #pragma unroll
    for (int i = 0; i < 2; ++i)
        #pragma unroll
        for (int j = 0; j < 4; ++j) {
            float bv = bias[colbase + j * 16 + r];
            int dd = j * 16 + r;
            #pragma unroll
            for (int v = 0; v < 8; ++v) {
                int t = t0 + wm * 32 + i * 16 + v + hi * 8;
                dst[(size_t)t * HD + dd] = (__bf16)(acc[i][j][v] + bv);
            }
        }
}

// ------------------------------------------------------------- attention core
// One WG per (b, h, 128 q-rows). 8 waves; wave w owns q rows qw..qw+15.
// Flash-style online softmax over 64-key blocks; all matmuls via WMMA bf16.
// K tile staged by the Tensor Data Mover (contiguous 8KB), V transposed by hand.
__global__ __launch_bounds__(256) void k_attn(
    const __bf16* __restrict__ Qh, const __bf16* __restrict__ Kh,
    const __bf16* __restrict__ Vh, __bf16* __restrict__ Y)
{
    __shared__ __bf16 Ks[64 * 64];      // [key][d]  (rows are B-frag columns)
    __shared__ __bf16 Vt[64 * 64];      // [d][key]  (transposed V)
    __shared__ __bf16 Ps[8][16 * 64];   // per-wave P staging
    const int tid = threadIdx.x, lane = tid & 31, wave = tid >> 5;
    const int r = lane & 15, hi = lane >> 4;
    const int qblk = blockIdx.x & 7;
    const int h    = (blockIdx.x >> 3) % NH;
    const int b    = blockIdx.x / (8 * NH);
    const size_t hb = ((size_t)(b * NH + h)) * TT * HD;
    const __bf16* qp = Qh + hb;
    const __bf16* kp = Kh + hb;
    const __bf16* vp = Vh + hb;
    const int qw = qblk * 128 + wave * 16;

    // Q A-fragments straight from global (row-major [T][64] is K-contiguous)
    v16bf qf[2];
    {
        const __bf16* qrow = qp + (size_t)(qw + r) * HD;
        #pragma unroll
        for (int s = 0; s < 2; ++s) {
            u32x4 a = *(const u32x4*)(qrow + s * 32 + hi * 8);
            u32x4 c = *(const u32x4*)(qrow + s * 32 + hi * 8 + 16);
            qf[s] = make_frag(a, c);
        }
    }

    v8f zero = {};
    v8f yacc[4];
    #pragma unroll
    for (int j = 0; j < 4; ++j) yacc[j] = zero;
    float mrow[8], lrow[8];
    #pragma unroll
    for (int v = 0; v < 8; ++v) { mrow[v] = -3.0e38f; lrow[v] = 0.f; }

    const unsigned ks_lds = (unsigned)(unsigned long long)(size_t)&Ks[0];

    const int nkb = (qblk + 1) * 2;     // 64-key blocks needed (causal)
    for (int kbi = 0; kbi < nkb; ++kbi) {
        const int kk0 = kbi * 64;
        if (wave == 0) {   // wave-uniform: TDM copy of contiguous K tile (8KB)
            unsigned long long ga =
                (unsigned long long)(size_t)(kp + (size_t)kk0 * HD);
            tdm_load_1d(ks_lds, ga, 64 * HD);
            asm volatile("s_wait_tensorcnt 0x0" ::: "memory");
        }
        #pragma unroll
        for (int t2 = 0; t2 < 2; ++t2) {  // V tile transposed: Vt[d][key]
            int tsk = tid + t2 * 256;
            int key = tsk & 63, dc = tsk >> 6;
            u32x4 dv = *(const u32x4*)(vp + (size_t)(kk0 + key) * HD + dc * 8);
            const __bf16* e = (const __bf16*)&dv;
            #pragma unroll
            for (int j = 0; j < 8; ++j) Vt[(dc * 8 + j) * 64 + key] = e[j];
        }
        __syncthreads();

        if (kk0 <= qw + 15) {  // wave-uniform: EXEC stays all-ones for WMMA
            // S = Q @ K^T   (4 n-tiles x 2 K-steps)
            v8f sacc[4];
            #pragma unroll
            for (int j = 0; j < 4; ++j) sacc[j] = zero;
            #pragma unroll
            for (int s = 0; s < 2; ++s)
                #pragma unroll
                for (int j = 0; j < 4; ++j) {
                    v16bf kf = load_frag(Ks + (j * 16) * 64, 64, s * 32);
                    sacc[j] = wmma_bf16(qf[s], kf, sacc[j]);
                }
            // scale + causal mask (C layout: row = v + 8*hi, col = lane&15)
            #pragma unroll
            for (int j = 0; j < 4; ++j)
                #pragma unroll
                for (int v = 0; v < 8; ++v) {
                    float xv = sacc[j][v] * 0.125f;     // 1/sqrt(64)
                    int mg = qw + v + hi * 8;
                    int ng = kk0 + j * 16 + r;
                    sacc[j][v] = (ng > mg) ? -3.0e38f : xv;
                }
            // online softmax: row max over 4 tiles + 16 lanes of the half-wave
            float scl[8];
            #pragma unroll
            for (int v = 0; v < 8; ++v) {
                float m = fmaxf(fmaxf(sacc[0][v], sacc[1][v]),
                                fmaxf(sacc[2][v], sacc[3][v]));
                #pragma unroll
                for (int d = 1; d < 16; d <<= 1) m = fmaxf(m, __shfl_xor(m, d, 32));
                float mn = fmaxf(mrow[v], m);
                scl[v] = __expf(mrow[v] - mn);
                mrow[v] = mn;
            }
            float rsum[8];
            #pragma unroll
            for (int v = 0; v < 8; ++v) rsum[v] = 0.f;
            #pragma unroll
            for (int j = 0; j < 4; ++j)
                #pragma unroll
                for (int v = 0; v < 8; ++v) {
                    float p = __expf(sacc[j][v] - mrow[v]);
                    sacc[j][v] = p;
                    rsum[v] += p;
                }
            #pragma unroll
            for (int v = 0; v < 8; ++v) {
                float s = rsum[v];
                #pragma unroll
                for (int d = 1; d < 16; d <<= 1) s += __shfl_xor(s, d, 32);
                lrow[v] = lrow[v] * scl[v] + s;
            }
            #pragma unroll
            for (int j = 0; j < 4; ++j)
                #pragma unroll
                for (int v = 0; v < 8; ++v) yacc[j][v] *= scl[v];
            // stage P (bf16) in per-wave LDS, reload as A-fragments
            __bf16* pl = &Ps[wave][0];
            #pragma unroll
            for (int j = 0; j < 4; ++j)
                #pragma unroll
                for (int v = 0; v < 8; ++v)
                    pl[(v + hi * 8) * 64 + j * 16 + r] = (__bf16)sacc[j][v];
            asm volatile("s_wait_dscnt 0" ::: "memory");
            // y += P @ V
            #pragma unroll
            for (int s = 0; s < 2; ++s) {
                v16bf pf = load_frag(pl, 64, s * 32);
                #pragma unroll
                for (int j = 0; j < 4; ++j) {
                    v16bf vf = load_frag(Vt + (j * 16) * 64, 64, s * 32);
                    yacc[j] = wmma_bf16(pf, vf, yacc[j]);
                }
            }
        }
        __syncthreads();
    }

    // normalize and store y (bf16, token-major [B,T,C] for the proj GEMM)
    #pragma unroll
    for (int v = 0; v < 8; ++v) {
        float inv = 1.0f / lrow[v];
        int t = qw + v + hi * 8;
        #pragma unroll
        for (int j = 0; j < 4; ++j) {
            int col = h * HD + j * 16 + r;
            Y[((size_t)b * TT + t) * EMB + col] = (__bf16)(yacc[j][v] * inv);
        }
    }
}

// ------------------------------------------------------------------ proj GEMM
// [8192,768](bf16) @ [768,768](bf16) + bias -> fp32 d_out
__global__ __launch_bounds__(256) void k_proj(
    const __bf16* __restrict__ X, const __bf16* __restrict__ W,
    const float* __restrict__ bias, float* __restrict__ out)
{
    __shared__ __bf16 As[128 * 32];
    __shared__ __bf16 Bs[128 * 32];
    const int tid  = threadIdx.x;
    const int lane = tid & 31, wave = tid >> 5;
    const int wm = wave & 3, wn = wave >> 2;
    const int bm0 = blockIdx.x * 128, bn0 = blockIdx.y * 128;
    const int r = lane & 15, hi = lane >> 4;

    v8f zero = {};
    v8f acc[2][4];
    #pragma unroll
    for (int i = 0; i < 2; ++i)
        #pragma unroll
        for (int j = 0; j < 4; ++j) acc[i][j] = zero;

    for (int k0 = 0; k0 < EMB; k0 += 32) {
        {
            int ar = tid >> 1, ah = tid & 1;
            const u32x4* g = (const u32x4*)(X + (size_t)(bm0 + ar) * EMB + k0 + ah * 16);
            u32x4 a0 = g[0], a1 = g[1];
            u32x4* sp = (u32x4*)(As + ar * 32 + ah * 16);
            sp[0] = a0; sp[1] = a1;
        }
        #pragma unroll
        for (int t2 = 0; t2 < 2; ++t2) {
            int tsk = tid + t2 * 256;
            int nc = tsk & 15, kr = tsk >> 4;
            u32x4 dv = *(const u32x4*)(W + (size_t)(k0 + kr) * EMB + bn0 + nc * 8);
            const __bf16* e = (const __bf16*)&dv;
            #pragma unroll
            for (int j = 0; j < 8; ++j)
                Bs[(nc * 8 + j) * 32 + kr] = e[j];
        }
        __syncthreads();
        v16bf af[2], bfr[4];
        #pragma unroll
        for (int i = 0; i < 2; ++i)
            af[i] = load_frag(As + (wm * 32 + i * 16) * 32, 32, 0);
        #pragma unroll
        for (int j = 0; j < 4; ++j)
            bfr[j] = load_frag(Bs + (wn * 64 + j * 16) * 32, 32, 0);
        #pragma unroll
        for (int i = 0; i < 2; ++i)
            #pragma unroll
            for (int j = 0; j < 4; ++j)
                acc[i][j] = wmma_bf16(af[i], bfr[j], acc[i][j]);
        __syncthreads();
    }

    #pragma unroll
    for (int i = 0; i < 2; ++i)
        #pragma unroll
        for (int j = 0; j < 4; ++j) {
            int ng = bn0 + wn * 64 + j * 16 + r;
            float bv = bias[ng];
            #pragma unroll
            for (int v = 0; v < 8; ++v) {
                int mg = bm0 + wm * 32 + i * 16 + v + hi * 8;
                out[(size_t)mg * EMB + ng] = acc[i][j][v] + bv;
            }
        }
}

// -------------------------------------------------------------------- launch
extern "C" void kernel_launch(void* const* d_in, const int* in_sizes, int n_in,
                              void* d_out, int out_size, void* d_ws, size_t ws_size,
                              hipStream_t stream) {
    const float* x  = (const float*)d_in[0];
    const float* Wa = (const float*)d_in[1];
    const float* ba = (const float*)d_in[2];
    const float* Wp = (const float*)d_in[3];
    const float* bp = (const float*)d_in[4];
    float* out = (float*)d_out;

    char* ws = (char*)d_ws;
    size_t off = 0;
    auto take = [&](size_t elems) -> __bf16* {
        __bf16* p = (__bf16*)(ws + off);
        off += ((elems * sizeof(__bf16)) + 255) & ~(size_t)255;
        return p;
    };
    __bf16* Xb  = take((size_t)NTOK * EMB);        // 12.6 MB
    __bf16* Wab = take((size_t)EMB * EMB3);        //  3.5 MB
    __bf16* Wpb = take((size_t)EMB * EMB);         //  1.2 MB
    __bf16* Qh  = take((size_t)BB * NH * TT * HD); // 12.6 MB
    __bf16* Kh  = take((size_t)BB * NH * TT * HD);
    __bf16* Vh  = take((size_t)BB * NH * TT * HD);
    __bf16* Yb  = take((size_t)NTOK * EMB);        // 12.6 MB
    (void)ws_size; (void)in_sizes; (void)n_in; (void)out_size;

    int n4;
    n4 = NTOK * EMB / 4;
    k_cvt<<<(n4 + 255) / 256, 256, 0, stream>>>(x, Xb, n4);
    n4 = EMB * EMB3 / 4;
    k_cvt<<<(n4 + 255) / 256, 256, 0, stream>>>(Wa, Wab, n4);
    n4 = EMB * EMB / 4;
    k_cvt<<<(n4 + 255) / 256, 256, 0, stream>>>(Wp, Wpb, n4);

    dim3 g1(NTOK / 128, EMB3 / 128);   // 64 x 18
    k_qkv<<<g1, 256, 0, stream>>>(Xb, Wab, ba, Qh, Kh, Vh);

    k_attn<<<BB * NH * (TT / 128), 256, 0, stream>>>(Qh, Kh, Vh, Yb);

    dim3 g2(NTOK / 128, EMB / 128);    // 64 x 6
    k_proj<<<g2, 256, 0, stream>>>(Yb, Wpb, bp, out);
}
